// iFairyAttention_19550691131411
// MI455X (gfx1250) — matrix operand order repaired and checked by reading, not compiled, not run on hardware
//
#include <hip/hip_runtime.h>
#include <hip/hip_bf16.h>

// ---------------------------------------------------------------------------
// iFairyAttention on MI455X (gfx1250): all contractions on
// v_wmma_f32_16x16x32_f16 (wave32 WMMA); f32 for softmax stats/accumulators.
// GEMMs use a register-staged software pipeline (global loads for tile k+1
// overlap the WMMA block of tile k); LDS staging uses wide b128 DS ops with
// in-register transposes.
// ---------------------------------------------------------------------------

typedef __attribute__((ext_vector_type(16))) _Float16 v16h;
typedef __attribute__((ext_vector_type(8)))  _Float16 v8h;
typedef __attribute__((ext_vector_type(8)))  float    v8f;
typedef __attribute__((ext_vector_type(4)))  float    v4f;

#define HID   2048
#define NHEAD 16
#define HDIM  128
#define SEQ   2048
#define QKVN  6144   // 2048 q + 2048 k + 2048 v (x r/i separately)

// ---- WMMA fragment loaders (layouts per CDNA5 ISA 7.12.2) -----------------
// A 16x32 f16: lane l -> row l&15; VGPR0-3 = K 8h..8h+7, VGPR4-7 = K 16+8h..
// (h = l>>4). p must point at rowBase + 8h; reads p[0..7] and p[16..23].
__device__ __forceinline__ v16h frag_a(const _Float16* p) {
  v8h lo = *(const v8h*)p;
  v8h hi = *(const v8h*)(p + 16);
  v16h r;
#pragma unroll
  for (int i = 0; i < 8; ++i) { r[i] = lo[i]; r[i + 8] = hi[i]; }
  return r;
}
// B 32x16 f16: lane l -> col l&15; K 8h..8h+7 then K 16+8h..16+8h+7.
// p points at colBase + 8h; reads p[0..7] and p[16..23] (two 16B DS loads).
__device__ __forceinline__ v16h frag_b(const _Float16* p) {
  v8h lo = *(const v8h*)p;
  v8h hi = *(const v8h*)(p + 16);
  v16h r;
#pragma unroll
  for (int i = 0; i < 8; ++i) { r[i] = lo[i]; r[i + 8] = hi[i]; }
  return r;
}

__device__ __forceinline__ v8f wmma_f16(v16h a, v16h b, v8f c) {
  return __builtin_amdgcn_wmma_f32_16x16x32_f16(false, a, false, b,
                                                (short)0, c, false, false);
}

// ---------------------------------------------------------------------------
// Pipelined tiled GEMM: C[M,N] = A[M,K]*B[K,N] (+bias). BM=BN=128, BK=64.
// 256 threads = 8 waves (2x4); each wave -> 64x32 (4x2 16x16 tiles),
// 16 WMMAs per k-step. Register-staged double issue: load tile k+1 to regs,
// barrier, 16 WMMAs on tile k from LDS, barrier, cvt+store regs to LDS.
// ---------------------------------------------------------------------------
template <bool A_HALF, bool HAS_BIAS, bool OUT_HALF>
__global__ __launch_bounds__(256) void gemm_wmma(
    const void* __restrict__ Av, const float* __restrict__ B,
    const float* __restrict__ bias, void* __restrict__ Cv,
    int M, int N, int K) {
  __shared__ _Float16 As[128 * 72];  // [row][k], stride 72 halfs
  __shared__ _Float16 Bs[128 * 72];  // transposed: [n][k], stride 72 halfs

  const int tid  = threadIdx.x;
  const int lane = tid & 31;
  const int wave = tid >> 5;
  const int wm   = wave >> 2;        // 0..1 -> 64 rows
  const int wn   = wave & 3;         // 0..3 -> 32 cols
  const int l15  = lane & 15;
  const int lh   = lane >> 4;
  const int bm0  = blockIdx.y * 128;
  const int bn0  = blockIdx.x * 128;

  // A staging: thread -> (row, 32-col segment)
  const int ar = tid >> 1, aseg = tid & 1;
  // B staging: thread -> 8(k) x 4(n) patch, transposed in registers
  const int bkg = tid >> 5;          // 0..7 -> k rows bkg*8..+7
  const int bng = tid & 31;          // 0..31 -> cols bng*4..+3

  v8f acc[4][2] = {};

  const int nk = K >> 6;

  // ---- staged tile registers ----
  v8h ah[4];        // A as f16 (A_HALF path)
  v4f af4[8];       // A as f32
  v4f bf4[8];       // B rows (f32)

  // prologue: load tile 0
  {
    if (A_HALF) {
      const _Float16* ga = (const _Float16*)Av + (size_t)(bm0 + ar) * K + aseg * 32;
#pragma unroll
      for (int j = 0; j < 4; ++j) ah[j] = *(const v8h*)(ga + j * 8);
    } else {
      const float* ga = (const float*)Av + (size_t)(bm0 + ar) * K + aseg * 32;
#pragma unroll
      for (int j = 0; j < 8; ++j) af4[j] = *(const v4f*)(ga + j * 4);
    }
    const float* gb = B + (size_t)(bkg * 8) * N + bn0 + bng * 4;
#pragma unroll
    for (int kk = 0; kk < 8; ++kk) bf4[kk] = *(const v4f*)(gb + (size_t)kk * N);
  }
  // store tile 0 (disjoint regions; no barrier needed before first store)
  {
    if (A_HALF) {
#pragma unroll
      for (int j = 0; j < 4; ++j)
        *(v8h*)&As[ar * 72 + aseg * 32 + j * 8] = ah[j];
    } else {
#pragma unroll
      for (int j = 0; j < 4; ++j) {
        v8h p;
#pragma unroll
        for (int e = 0; e < 4; ++e) {
          p[e]     = (_Float16)af4[2 * j][e];
          p[e + 4] = (_Float16)af4[2 * j + 1][e];
        }
        *(v8h*)&As[ar * 72 + aseg * 32 + j * 8] = p;
      }
    }
#pragma unroll
    for (int nn = 0; nn < 4; ++nn) {
      v8h p;
#pragma unroll
      for (int kk = 0; kk < 8; ++kk) p[kk] = (_Float16)bf4[kk][nn];
      *(v8h*)&Bs[(bng * 4 + nn) * 72 + bkg * 8] = p;
    }
  }

  for (int kb = 0; kb < nk; ++kb) {
    const bool has_next = (kb + 1) < nk;
    // ---- issue global loads for tile kb+1 (latency hidden behind WMMAs) ----
    if (has_next) {
      const int k0 = (kb + 1) << 6;
      if (A_HALF) {
        const _Float16* ga =
            (const _Float16*)Av + (size_t)(bm0 + ar) * K + k0 + aseg * 32;
#pragma unroll
        for (int j = 0; j < 4; ++j) ah[j] = *(const v8h*)(ga + j * 8);
      } else {
        const float* ga =
            (const float*)Av + (size_t)(bm0 + ar) * K + k0 + aseg * 32;
#pragma unroll
        for (int j = 0; j < 8; ++j) af4[j] = *(const v4f*)(ga + j * 4);
        __builtin_prefetch(ga + 64, 0, 1);
      }
      const float* gb = B + (size_t)(k0 + bkg * 8) * N + bn0 + bng * 4;
#pragma unroll
      for (int kk = 0; kk < 8; ++kk)
        bf4[kk] = *(const v4f*)(gb + (size_t)kk * N);
      __builtin_prefetch(gb + (size_t)64 * N, 0, 1);
    }
    __syncthreads();  // tile kb now visible in LDS

    // ---- 16 WMMAs on tile kb ----
#pragma unroll
    for (int c = 0; c < 2; ++c) {
      v16h af[4];
#pragma unroll
      for (int mi = 0; mi < 4; ++mi)
        af[mi] = frag_a(&As[(wm * 64 + mi * 16 + l15) * 72 + c * 32 + lh * 8]);
#pragma unroll
      for (int nj = 0; nj < 2; ++nj) {
        v16h bf = frag_b(&Bs[(wn * 32 + nj * 16 + l15) * 72 + c * 32 + lh * 8]);
#pragma unroll
        for (int mi = 0; mi < 4; ++mi)
          acc[mi][nj] = wmma_f16(af[mi], bf, acc[mi][nj]);
      }
    }
    __syncthreads();  // LDS reads done before overwrite

    // ---- cvt + store staged tile kb+1 ----
    if (has_next) {
      if (A_HALF) {
#pragma unroll
        for (int j = 0; j < 4; ++j)
          *(v8h*)&As[ar * 72 + aseg * 32 + j * 8] = ah[j];
      } else {
#pragma unroll
        for (int j = 0; j < 4; ++j) {
          v8h p;
#pragma unroll
          for (int e = 0; e < 4; ++e) {
            p[e]     = (_Float16)af4[2 * j][e];
            p[e + 4] = (_Float16)af4[2 * j + 1][e];
          }
          *(v8h*)&As[ar * 72 + aseg * 32 + j * 8] = p;
        }
      }
#pragma unroll
      for (int nn = 0; nn < 4; ++nn) {
        v8h p;
#pragma unroll
        for (int kk = 0; kk < 8; ++kk) p[kk] = (_Float16)bf4[kk][nn];
        *(v8h*)&Bs[(bng * 4 + nn) * 72 + bkg * 8] = p;
      }
    }
  }

  // ---- epilogue: C layout M = v + 8*lh, N = l15 ----
#pragma unroll
  for (int nj = 0; nj < 2; ++nj) {
    const int col = bn0 + wn * 32 + nj * 16 + l15;
    const float bv = HAS_BIAS ? bias[col] : 0.0f;
#pragma unroll
    for (int mi = 0; mi < 4; ++mi) {
#pragma unroll
      for (int v = 0; v < 8; ++v) {
        const int row = bm0 + wm * 64 + mi * 16 + v + 8 * lh;
        const float x = acc[mi][nj][v] + bv;
        if (OUT_HALF)
          ((_Float16*)Cv)[(size_t)row * N + col] = (_Float16)x;
        else
          ((float*)Cv)[(size_t)row * N + col] = x;
      }
    }
  }
}

// ---------------------------------------------------------------------------
// RoPE over full head_dim on q (cols 0..2047) and k (cols 2048..4095) of the
// [S, 6144] f16 qkv buffers. Folds score scale 1/sqrt(D) into q.
// ---------------------------------------------------------------------------
__global__ __launch_bounds__(256) void rope_kernel(
    _Float16* __restrict__ qkv_r, _Float16* __restrict__ qkv_i,
    const int* __restrict__ positions, int total) {
  int idx = blockIdx.x * blockDim.x + threadIdx.x;
  if (idx >= total) return;
  const int srow = idx >> 12;        // / 4096
  const int c    = idx & 4095;       // 0..2047 q, 2048..4095 k
  const int d    = c & 127;
  const float pos  = (float)positions[srow];
  const float freq = __powf(1.0e6f, -(float)d * (1.0f / 128.0f));
  float sn, cs;
  __sincosf(pos * freq, &sn, &cs);
  const float scl = (c < 2048) ? 0.08838834764831845f : 1.0f;  // D^-0.5 into q
  const size_t off = (size_t)srow * QKVN + c;
  const float xr = (float)qkv_r[off], xi = (float)qkv_i[off];
  qkv_r[off] = (_Float16)((xr * cs - xi * sn) * scl);
  qkv_i[off] = (_Float16)((xr * sn + xi * cs) * scl);
}

// ---------------------------------------------------------------------------
// Causal flash attention, complex scores = q_r.k_r + q_i.k_i (scale folded
// into q). grid = (S/128, NHEAD); 8 waves x 16 query rows per block.
// Each kb step streams a 32-key K/V tile (r+i) through LDS; V is transposed
// in registers (8x8 patches) and stored with ds_store_b128.
// ---------------------------------------------------------------------------
__global__ __launch_bounds__(256) void attn_kernel(
    const _Float16* __restrict__ qkv_r, const _Float16* __restrict__ qkv_i,
    _Float16* __restrict__ out_r, _Float16* __restrict__ out_i) {
  __shared__ _Float16 Kr[32 * 136];   // [kv][d]
  __shared__ _Float16 Ki[32 * 136];
  __shared__ _Float16 Vr[128 * 40];   // transposed [d][kv] for PV B-frags
  __shared__ _Float16 Vi[128 * 40];
  __shared__ _Float16 Ps[8 * 16 * 40];  // per-wave P repack patch [m][kv]

  const int tid  = threadIdx.x;
  const int lane = tid & 31;
  const int wave = tid >> 5;
  const int l15  = lane & 15;
  const int lh   = lane >> 4;
  const int h    = blockIdx.y;
  const int q0   = blockIdx.x * 128;
  const int qrow = q0 + wave * 16;

  const int tsr = tid >> 6;   // 0:Kr 1:Ki 2:Vr 3:Vi
  const int t6  = tid & 63;

  // ---- Q fragments (r and i), 4 K-chunks of 32 over head_dim=128 ----
  v16h aqr[4], aqi[4];
  {
    const size_t base = (size_t)(qrow + l15) * QKVN + h * HDIM + lh * 8;
    const _Float16* pr = qkv_r + base;
    const _Float16* pi = qkv_i + base;
#pragma unroll
    for (int c = 0; c < 4; ++c) {
      aqr[c] = frag_a(pr + c * 32);
      aqi[c] = frag_a(pi + c * 32);
    }
  }

  v8f o_r[8] = {}, o_i[8] = {};
  float mst[8], lst[8];
#pragma unroll
  for (int v = 0; v < 8; ++v) { mst[v] = -3.0e38f; lst[v] = 0.0f; }

  const int kb_end = (q0 + 128) >> 5;
  for (int kb = 0; kb < kb_end; ++kb) {
    const int k0 = kb * 32;
    __syncthreads();  // prior iteration's LDS reads complete
    // ---- stage K (row-major, b128 copies) and V (d-major, reg transpose) ----
    if (tsr < 2) {
      const _Float16* src = (tsr == 1) ? qkv_i : qkv_r;
      const int kvrow = t6 >> 1, seg = t6 & 1;
      const _Float16* gp =
          src + (size_t)(k0 + kvrow) * QKVN + 2048 + h * HDIM + seg * 64;
      _Float16* dst = (tsr == 0 ? Kr : Ki) + kvrow * 136 + seg * 64;
#pragma unroll
      for (int j = 0; j < 8; ++j)
        *(v8h*)(dst + j * 8) = *(const v8h*)(gp + j * 8);
    } else {
      const _Float16* src = (tsr == 3) ? qkv_i : qkv_r;
      const int kv0 = (t6 >> 4) * 8;   // 0,8,16,24
      const int d0  = (t6 & 15) * 8;   // 0..120
      const _Float16* gp =
          src + (size_t)(k0 + kv0) * QKVN + 4096 + h * HDIM + d0;
      v8h rows[8];
#pragma unroll
      for (int e = 0; e < 8; ++e)
        rows[e] = *(const v8h*)(gp + (size_t)e * QKVN);
      _Float16* dst = (tsr == 2 ? Vr : Vi);
#pragma unroll
      for (int dd = 0; dd < 8; ++dd) {
        v8h col;
#pragma unroll
        for (int e = 0; e < 8; ++e) col[e] = rows[e][dd];
        *(v8h*)&dst[(d0 + dd) * 40 + kv0] = col;
      }
    }
    __syncthreads();

    const bool active = (k0 <= qrow + 15);
    v8f sc[2] = {};
    if (active) {
      // ---- scores: 16x32, Re(q conj(k)) = qr.kr + qi.ki ----
#pragma unroll
      for (int nj = 0; nj < 2; ++nj) {
#pragma unroll
        for (int c = 0; c < 4; ++c) {
          v16h bkr = frag_b(&Kr[(nj * 16 + l15) * 136 + c * 32 + lh * 8]);
          v16h bki = frag_b(&Ki[(nj * 16 + l15) * 136 + c * 32 + lh * 8]);
          sc[nj] = wmma_f16(aqr[c], bkr, sc[nj]);
          sc[nj] = wmma_f16(aqi[c], bki, sc[nj]);
        }
      }
      // ---- causal mask ----
#pragma unroll
      for (int nj = 0; nj < 2; ++nj) {
        const int key = k0 + nj * 16 + l15;
#pragma unroll
        for (int v = 0; v < 8; ++v) {
          const int query = qrow + v + 8 * lh;
          if (key > query) sc[nj][v] = -1.0e30f;
        }
      }
      // ---- online softmax (row stats replicated across 16 N-lanes) ----
#pragma unroll
      for (int v = 0; v < 8; ++v) {
        float mx = fmaxf(sc[0][v], sc[1][v]);
#pragma unroll
        for (int off = 1; off < 16; off <<= 1)
          mx = fmaxf(mx, __shfl_xor(mx, off, 32));
        const float mnew = fmaxf(mst[v], mx);
        const float corr = __expf(mst[v] - mnew);
        const float p0 = __expf(sc[0][v] - mnew);
        const float p1 = __expf(sc[1][v] - mnew);
        sc[0][v] = p0;
        sc[1][v] = p1;
        float rs = p0 + p1;
#pragma unroll
        for (int off = 1; off < 16; off <<= 1)
          rs += __shfl_xor(rs, off, 32);
        lst[v] = lst[v] * corr + rs;
        mst[v] = mnew;
#pragma unroll
        for (int dt = 0; dt < 8; ++dt) {
          o_r[dt][v] *= corr;
          o_i[dt][v] *= corr;
        }
      }
      // ---- repack P: C-layout -> LDS row-major [m][kv] (own wave patch) ----
#pragma unroll
      for (int nj = 0; nj < 2; ++nj)
#pragma unroll
        for (int v = 0; v < 8; ++v)
          Ps[wave * 640 + (v + 8 * lh) * 40 + nj * 16 + l15] =
              (_Float16)sc[nj][v];
    }
    __syncthreads();  // uniform; also orders P store -> P load
    if (active) {
      v16h ap = frag_a(&Ps[wave * 640 + l15 * 40 + lh * 8]);
#pragma unroll
      for (int dt = 0; dt < 8; ++dt) {
        v16h bvr = frag_b(&Vr[(dt * 16 + l15) * 40 + lh * 8]);
        v16h bvi = frag_b(&Vi[(dt * 16 + l15) * 40 + lh * 8]);
        o_r[dt] = wmma_f16(ap, bvr, o_r[dt]);
        o_i[dt] = wmma_f16(ap, bvi, o_i[dt]);
      }
    }
  }

  // ---- normalize + store f16 attention output [S, H*D] ----
#pragma unroll
  for (int v = 0; v < 8; ++v) {
    const float inv = 1.0f / lst[v];
    const int row = qrow + v + 8 * lh;
#pragma unroll
    for (int dt = 0; dt < 8; ++dt) {
      const size_t o = (size_t)row * (NHEAD * HDIM) + h * HDIM + dt * 16 + l15;
      out_r[o] = (_Float16)(o_r[dt][v] * inv);
      out_i[o] = (_Float16)(o_i[dt][v] * inv);
    }
  }
}

// ---------------------------------------------------------------------------
extern "C" void kernel_launch(void* const* d_in, const int* in_sizes, int n_in,
                              void* d_out, int out_size, void* d_ws,
                              size_t ws_size, hipStream_t stream) {
  (void)in_sizes; (void)n_in; (void)out_size; (void)ws_size;
  const float* hidden    = (const float*)d_in[0];
  const int*   positions = (const int*)d_in[1];
  const float* Wqkv_r    = (const float*)d_in[2];
  const float* bqkv_r    = (const float*)d_in[3];
  const float* Wqkv_i    = (const float*)d_in[4];
  const float* bqkv_i    = (const float*)d_in[5];
  const float* Wo_r      = (const float*)d_in[6];
  const float* Wo_i      = (const float*)d_in[7];
  float* out = (float*)d_out;

  char* ws = (char*)d_ws;
  _Float16* qkvr = (_Float16*)(ws);                              // S*6144 f16
  _Float16* qkvi = (_Float16*)(ws + (size_t)SEQ * QKVN * 2);     // S*6144 f16
  _Float16* aor  = (_Float16*)(ws + (size_t)SEQ * QKVN * 4);     // S*2048 f16
  _Float16* aoi  = (_Float16*)(ws + (size_t)SEQ * QKVN * 4 +
                               (size_t)SEQ * HID * 2);           // S*2048 f16

  dim3 blk(256);

  // 1) dual QKV projections (f16 out, +bias)
  dim3 g1(QKVN / 128, SEQ / 128);
  gemm_wmma<false, true, true><<<g1, blk, 0, stream>>>(
      hidden, Wqkv_r, bqkv_r, qkvr, SEQ, QKVN, HID);
  gemm_wmma<false, true, true><<<g1, blk, 0, stream>>>(
      hidden, Wqkv_i, bqkv_i, qkvi, SEQ, QKVN, HID);

  // 2) complex RoPE on q,k (scale folded into q)
  const int nrope = SEQ * 4096;
  rope_kernel<<<(nrope + 255) / 256, blk, 0, stream>>>(qkvr, qkvi, positions,
                                                       nrope);

  // 3) causal complex flash attention
  dim3 ga(SEQ / 128, NHEAD);
  attn_kernel<<<ga, blk, 0, stream>>>(qkvr, qkvi, aor, aoi);

  // 4) output projections (f32 out to d_out, concat [out_r | out_i])
  dim3 g2(HID / 128, SEQ / 128);
  gemm_wmma<true, false, false><<<g2, blk, 0, stream>>>(
      aor, Wo_r, nullptr, out, SEQ, HID, HID);
  gemm_wmma<true, false, false><<<g2, blk, 0, stream>>>(
      aoi, Wo_i, nullptr, out + (size_t)SEQ * HID, SEQ, HID, HID);
}
